// AdaptiveSoftmaxProductHead_65790309040722
// MI455X (gfx1250) — matrix operand order repaired and checked by reading, not compiled, not run on hardware
//
#include <hip/hip_runtime.h>
#include <hip/hip_bf16.h>

typedef __bf16 bf16_t;
typedef __attribute__((ext_vector_type(16))) __bf16 v16bf;
typedef __attribute__((ext_vector_type(8)))  __bf16 v8bf;
typedef __attribute__((ext_vector_type(8)))  float  v8f;

#define NEG_INF (-__builtin_inff())

// ---------------------------------------------------------------------------
// f32 -> bf16 convert with zero padding (srcR x srcC) -> (dstR x dstC)
// ---------------------------------------------------------------------------
__global__ void cvt_pad_kernel(const float* __restrict__ src, bf16_t* __restrict__ dst,
                               int srcR, int srcC, int dstR, int dstC) {
  long total = (long)dstR * dstC;
  for (long i = blockIdx.x * (long)blockDim.x + threadIdx.x; i < total;
       i += (long)gridDim.x * blockDim.x) {
    int r = (int)(i / dstC);
    int c = (int)(i - (long)r * dstC);
    float v = (r < srcR && c < srcC) ? src[(long)r * srcC + c] : 0.0f;
    dst[i] = (bf16_t)v;
  }
}

// ---------------------------------------------------------------------------
// WMMA fragment loaders per CDNA5 ISA 7.12.2 (16-bit A 16x32, B 32x16, wave32)
// A row-major [rows x Kpad]; lane l holds row (rowBase + l%16),
//   elems 0..7 -> K = k0 + (l>=16?8:0) + e ; elems 8..15 -> +16.
// B class-major [cols x Kpad] (logits = A x B^T); lane l holds class col
//   (colBase + l%16), elems 0..7 -> K = k0 + (l>=16?8:0) + e ; elems 8..15 -> +16.
// ---------------------------------------------------------------------------
__device__ __forceinline__ v16bf load_a_frag(const bf16_t* __restrict__ A, int Kpad,
                                             int rowBase, int lane, int k0) {
  int lo = lane & 15, hi = lane >> 4;
  const bf16_t* base = A + (long)(rowBase + lo) * Kpad + k0 + (hi ? 8 : 0);
  v8bf alo = *(const v8bf*)(base);
  v8bf ahi = *(const v8bf*)(base + 16);
  v16bf a;
#pragma unroll
  for (int e = 0; e < 8; ++e) { a[e] = alo[e]; a[8 + e] = ahi[e]; }
  return a;
}

__device__ __forceinline__ v16bf load_b_frag(const bf16_t* __restrict__ B, int Kpad,
                                             int colBase, int lane, int k0) {
  int lo = lane & 15, hi = lane >> 4;
  const bf16_t* base = B + (long)(colBase + lo) * Kpad + k0 + (hi ? 8 : 0);
  v8bf blo = *(const v8bf*)(base);
  v8bf bhi = *(const v8bf*)(base + 16);
  v16bf b;
#pragma unroll
  for (int e = 0; e < 8; ++e) { b[e] = blo[e]; b[8 + e] = bhi[e]; }
  return b;
}

// ---------------------------------------------------------------------------
// C[M x N] = A[M x K] * B[N x K]^T  (bf16 in, f32 out). One wave per 16x16 tile.
// grid = (M/16, N/16), block = 32.
// ---------------------------------------------------------------------------
__global__ void gemm16_bf16_nt(const bf16_t* __restrict__ A, const bf16_t* __restrict__ B,
                               float* __restrict__ C, int K, int N) {
  int lane = threadIdx.x;
  int rowBase = blockIdx.x * 16, colBase = blockIdx.y * 16;
  v8f c = {};
  for (int k0 = 0; k0 < K; k0 += 32) {
    v16bf a = load_a_frag(A, K, rowBase, lane, k0);
    v16bf b = load_b_frag(B, K, colBase, lane, k0);
    c = __builtin_amdgcn_wmma_f32_16x16x32_bf16(false, a, false, b, (short)0, c,
                                                false, false);
  }
  int lo = lane & 15, hi = lane >> 4;
#pragma unroll
  for (int r = 0; r < 8; ++r)
    C[(long)(rowBase + r + 8 * hi) * N + colBase + lo] = c[r];
}

// ---------------------------------------------------------------------------
// Fused logits + online logsumexp. Each wave owns 16 rows and a chunk of
// 16-wide column tiles. Streams tiles: WMMA -> per-lane running (max,sumexp)
// per row; captures target-column logit and optional fixed "extra" columns
// (head cluster logits). Cross-lane merge via LDS; writes split partials.
// grid = (nRows/16, S), block = 32.
// ---------------------------------------------------------------------------
__global__ void fused_lse_kernel(const bf16_t* __restrict__ A, const bf16_t* __restrict__ B,
                                 int Kpad, int nTiles, int chunkTiles, int validCols,
                                 const int* __restrict__ targets, int tlow, int thigh,
                                 float* __restrict__ pm, float* __restrict__ ps, int S,
                                 float* __restrict__ tgtbuf,
                                 float* __restrict__ extraBuf, int extraStart, int extraCount,
                                 int nRows) {
  __shared__ float shm[8 * 32];
  __shared__ float shs[8 * 32];
  int lane = threadIdx.x, lo = lane & 15, hi = lane >> 4;
  int rowBase = blockIdx.x * 16;
  int split = blockIdx.y;

  int trel[8];
#pragma unroll
  for (int r = 0; r < 8; ++r) {
    int t = targets[rowBase + r + 8 * hi];
    trel[r] = (t >= tlow && t < thigh) ? (t - tlow) : -1;
  }

  float m[8], s[8];
#pragma unroll
  for (int r = 0; r < 8; ++r) { m[r] = NEG_INF; s[r] = 0.0f; }

  int t0 = split * chunkTiles;
  int t1 = t0 + chunkTiles;
  if (t1 > nTiles) t1 = nTiles;

  for (int tile = t0; tile < t1; ++tile) {
    int colBase = tile * 16;
    v8f c = {};
    for (int k0 = 0; k0 < Kpad; k0 += 32) {
      v16bf a = load_a_frag(A, Kpad, rowBase, lane, k0);
      v16bf b = load_b_frag(B, Kpad, colBase, lane, k0);
      c = __builtin_amdgcn_wmma_f32_16x16x32_bf16(false, a, false, b, (short)0, c,
                                                  false, false);
    }
    int col = colBase + lo;
    if (col < validCols) {
#pragma unroll
      for (int r = 0; r < 8; ++r) {
        float v = c[r];
        if (v > m[r]) {            // rescale running sum
          s[r] = s[r] * __expf(m[r] - v) + 1.0f;  // exp(-inf)=0 on first hit
          m[r] = v;
        } else {
          s[r] += __expf(v - m[r]);
        }
        if (trel[r] == col) tgtbuf[rowBase + r + 8 * hi] = v;
      }
      if (extraCount > 0 && col >= extraStart && col < extraStart + extraCount) {
#pragma unroll
        for (int r = 0; r < 8; ++r)
          extraBuf[(long)(col - extraStart) * nRows + rowBase + r + 8 * hi] = c[r];
      }
    }
  }

#pragma unroll
  for (int r = 0; r < 8; ++r) { shm[r * 32 + lane] = m[r]; shs[r * 32 + lane] = s[r]; }
  __syncthreads();

  if (lane < 16) {
    int grp = (lane < 8) ? 0 : 16;   // rows 0..7 live in lanes 0..15, rows 8..15 in 16..31
    int r = lane & 7;
    float M = NEG_INF;
#pragma unroll
    for (int l = 0; l < 16; ++l) M = fmaxf(M, shm[r * 32 + grp + l]);
    float Ssum = 0.0f;
    if (M > NEG_INF) {
#pragma unroll
      for (int l = 0; l < 16; ++l) {
        float ml = shm[r * 32 + grp + l];
        if (ml > NEG_INF) Ssum += shs[r * 32 + grp + l] * __expf(ml - M);
      }
    }
    int row = rowBase + lane;
    pm[(long)row * S + split] = M;
    ps[(long)row * S + split] = Ssum;
  }
}

// ---------------------------------------------------------------------------
// Merge split partials into logsumexp and assemble the final output.
// ---------------------------------------------------------------------------
__device__ __forceinline__ float lse_reduce(const float* __restrict__ pm,
                                            const float* __restrict__ ps,
                                            int S, int row) {
  float M = NEG_INF;
  for (int j = 0; j < S; ++j) M = fmaxf(M, pm[(long)row * S + j]);
  float Ssum = 0.0f;
  for (int j = 0; j < S; ++j) {
    float ml = pm[(long)row * S + j];
    if (ml > NEG_INF) Ssum += ps[(long)row * S + j] * __expf(ml - M);
  }
  return M + __logf(Ssum);
}

__global__ void combine_kernel(const int* __restrict__ targets,
                               const float* pmh, const float* psh, int Sh,
                               const float* pm0, const float* ps0, int S0,
                               const float* pm1, const float* ps1, int S1,
                               const float* pm2, const float* ps2, int S2,
                               const float* __restrict__ tgtbuf,
                               const float* __restrict__ extraBuf,
                               float* __restrict__ out, int nRows) {
  int row = blockIdx.x * blockDim.x + threadIdx.x;
  if (row >= nRows) return;
  int t = targets[row];
  float lse_h = lse_reduce(pmh, psh, Sh, row);
  float res;
  if (t < 1000) {
    res = tgtbuf[row] - lse_h;                   // head shortlist log-prob
  } else {
    int ci; const float *pmt, *pst; int St;
    if (t < 10000)      { ci = 0; pmt = pm0; pst = ps0; St = S0; }
    else if (t < 50000) { ci = 1; pmt = pm1; pst = ps1; St = S1; }
    else                { ci = 2; pmt = pm2; pst = ps2; St = S2; }
    float lse_t = lse_reduce(pmt, pst, St, row);
    res = (extraBuf[(long)ci * nRows + row] - lse_h) + (tgtbuf[row] - lse_t);
  }
  out[row] = res;
}

// ---------------------------------------------------------------------------
// Host side
// ---------------------------------------------------------------------------
static inline int cvt_blocks(long total) {
  long b = (total + 255) / 256;
  return (int)(b > 4096 ? 4096 : b);
}

extern "C" void kernel_launch(void* const* d_in, const int* in_sizes, int n_in,
                              void* d_out, int out_size, void* d_ws, size_t ws_size,
                              hipStream_t stream) {
  (void)in_sizes; (void)n_in; (void)out_size; (void)ws_size;
  const float* x       = (const float*)d_in[0];
  const int*   targets = (const int*)  d_in[1];
  const float* head_w  = (const float*)d_in[2];
  const float* w1_0    = (const float*)d_in[3];
  const float* w2_0    = (const float*)d_in[4];
  const float* w1_1    = (const float*)d_in[5];
  const float* w2_1    = (const float*)d_in[6];
  const float* w1_2    = (const float*)d_in[7];
  const float* w2_2    = (const float*)d_in[8];
  float* out = (float*)d_out;

  const int N = 2048, D = 512;
  const int HCOLS = 1008;       // 1003 padded to 16
  const int OSZ0 = 9000, OSZ0P = 9008, K0 = 128;
  const int OSZ1 = 40000, K1 = 32;
  const int OSZ2 = 50000, K2raw = 8, K2 = 32;
  const int Sh = 4, S0 = 5, S1 = 20, S2 = 25;

  char* ws = (char*)d_ws;
  size_t off = 0;
  auto alloc = [&](size_t bytes) -> void* {
    void* p = ws + off;
    off = (off + bytes + 255) & ~(size_t)255;
    return p;
  };

  bf16_t* x_bf   = (bf16_t*)alloc((size_t)N * D * 2);
  bf16_t* hw_bf  = (bf16_t*)alloc((size_t)HCOLS * D * 2);
  bf16_t* w10_bf = (bf16_t*)alloc((size_t)128 * D * 2);
  bf16_t* w11_bf = (bf16_t*)alloc((size_t)32 * D * 2);
  bf16_t* w12_bf = (bf16_t*)alloc((size_t)16 * D * 2);
  bf16_t* w20_bf = (bf16_t*)alloc((size_t)OSZ0P * K0 * 2);
  bf16_t* w21_bf = (bf16_t*)alloc((size_t)OSZ1 * K1 * 2);
  bf16_t* w22_bf = (bf16_t*)alloc((size_t)OSZ2 * K2 * 2);
  float*  H0f    = (float*) alloc((size_t)N * 128 * 4);
  float*  H1f    = (float*) alloc((size_t)N * 32 * 4);
  float*  H2f    = (float*) alloc((size_t)N * 16 * 4);
  bf16_t* H0bf   = (bf16_t*)alloc((size_t)N * 128 * 2);
  bf16_t* H1bf   = (bf16_t*)alloc((size_t)N * 32 * 2);
  bf16_t* H2bf   = (bf16_t*)alloc((size_t)N * K2 * 2);
  float* pmh = (float*)alloc((size_t)N * Sh * 4);
  float* psh = (float*)alloc((size_t)N * Sh * 4);
  float* pm0 = (float*)alloc((size_t)N * S0 * 4);
  float* ps0 = (float*)alloc((size_t)N * S0 * 4);
  float* pm1 = (float*)alloc((size_t)N * S1 * 4);
  float* ps1 = (float*)alloc((size_t)N * S1 * 4);
  float* pm2 = (float*)alloc((size_t)N * S2 * 4);
  float* ps2 = (float*)alloc((size_t)N * S2 * 4);
  float* tgtbuf = (float*)alloc((size_t)N * 4);
  float* extra  = (float*)alloc((size_t)3 * N * 4);

  // --- bf16 conversion / padding ---
  cvt_pad_kernel<<<cvt_blocks((long)N * D), 256, 0, stream>>>(x, x_bf, N, D, N, D);
  cvt_pad_kernel<<<cvt_blocks((long)HCOLS * D), 256, 0, stream>>>(head_w, hw_bf, 1003, D, HCOLS, D);
  cvt_pad_kernel<<<cvt_blocks((long)128 * D), 256, 0, stream>>>(w1_0, w10_bf, 128, D, 128, D);
  cvt_pad_kernel<<<cvt_blocks((long)32 * D), 256, 0, stream>>>(w1_1, w11_bf, 32, D, 32, D);
  cvt_pad_kernel<<<cvt_blocks((long)16 * D), 256, 0, stream>>>(w1_2, w12_bf, 8, D, 16, D);
  cvt_pad_kernel<<<cvt_blocks((long)OSZ0P * K0), 256, 0, stream>>>(w2_0, w20_bf, OSZ0, K0, OSZ0P, K0);
  cvt_pad_kernel<<<cvt_blocks((long)OSZ1 * K1), 256, 0, stream>>>(w2_1, w21_bf, OSZ1, K1, OSZ1, K1);
  cvt_pad_kernel<<<cvt_blocks((long)OSZ2 * K2), 256, 0, stream>>>(w2_2, w22_bf, OSZ2, K2raw, OSZ2, K2);

  // --- tail projections H_i = x @ w1_i^T (WMMA) ---
  gemm16_bf16_nt<<<dim3(N / 16, 128 / 16), 32, 0, stream>>>(x_bf, w10_bf, H0f, D, 128);
  gemm16_bf16_nt<<<dim3(N / 16, 32 / 16), 32, 0, stream>>>(x_bf, w11_bf, H1f, D, 32);
  gemm16_bf16_nt<<<dim3(N / 16, 16 / 16), 32, 0, stream>>>(x_bf, w12_bf, H2f, D, 16);

  cvt_pad_kernel<<<cvt_blocks((long)N * 128), 256, 0, stream>>>(H0f, H0bf, N, 128, N, 128);
  cvt_pad_kernel<<<cvt_blocks((long)N * 32), 256, 0, stream>>>(H1f, H1bf, N, 32, N, 32);
  cvt_pad_kernel<<<cvt_blocks((long)N * K2), 256, 0, stream>>>(H2f, H2bf, N, 16, N, K2);

  // --- fused logits + online LSE ---
  // head: 63 tiles (1008 cols), chunk 16 -> S=4 splits; capture target (t<1000)
  // and cluster logit columns 1000..1002.
  fused_lse_kernel<<<dim3(N / 16, Sh), 32, 0, stream>>>(
      x_bf, hw_bf, D, HCOLS / 16, 16, 1003, targets, 0, 1000,
      pmh, psh, Sh, tgtbuf, extra, 1000, 3, N);
  // cluster 0: 563 tiles, chunk 128 -> S=5
  fused_lse_kernel<<<dim3(N / 16, S0), 32, 0, stream>>>(
      H0bf, w20_bf, K0, OSZ0P / 16, 128, OSZ0, targets, 1000, 10000,
      pm0, ps0, S0, tgtbuf, nullptr, 0, 0, N);
  // cluster 1: 2500 tiles, chunk 125 -> S=20
  fused_lse_kernel<<<dim3(N / 16, S1), 32, 0, stream>>>(
      H1bf, w21_bf, K1, OSZ1 / 16, 125, OSZ1, targets, 10000, 50000,
      pm1, ps1, S1, tgtbuf, nullptr, 0, 0, N);
  // cluster 2: 3125 tiles, chunk 125 -> S=25
  fused_lse_kernel<<<dim3(N / 16, S2), 32, 0, stream>>>(
      H2bf, w22_bf, K2, OSZ2 / 16, 125, OSZ2, targets, 50000, 100000,
      pm2, ps2, S2, tgtbuf, nullptr, 0, 0, N);

  // --- final assembly ---
  combine_kernel<<<dim3((N + 255) / 256), 256, 0, stream>>>(
      targets, pmh, psh, Sh, pm0, ps0, S0, pm1, ps1, S1, pm2, ps2, S2,
      tgtbuf, extra, out, N);
}
